// MoEByFunctionalGroup_42683384988082
// MI455X (gfx1250) — compile-verified, matrix operand order checked by reading, not run on hardware
//
#include <hip/hip_runtime.h>

// MoE (4 MLP + 4 KAN experts) for MI455X / gfx1250.
// Heavy contractions via v_wmma_f32_16x16x32_bf16; tiles staged into LDS by
// the Tensor Data Mover (tensor_load_to_lds + s_wait_tensorcnt) when the
// builtin is available, double-buffered, one barrier per K-step.

typedef __bf16 bf16_t;
typedef __attribute__((ext_vector_type(16))) __bf16 v16bf;
typedef __attribute__((ext_vector_type(8)))  __bf16 bf16x8;
typedef __attribute__((ext_vector_type(8)))  float  v8f;
typedef __attribute__((ext_vector_type(4)))  unsigned int u32x4;
typedef __attribute__((ext_vector_type(8)))  int i32x8;
typedef __attribute__((ext_vector_type(4)))  int i32x4;

#define B_ 4096
#define D_ 512
#define H_ 2048
#define O_ 512
#define NE_ 8

#if defined(__has_builtin)
#if __has_builtin(__builtin_amdgcn_tensor_load_to_lds) && \
    __has_builtin(__builtin_amdgcn_s_wait_tensorcnt)
#define USE_TDM 1
#endif
#endif

__device__ __forceinline__ bf16_t f2bf(float f) {
  unsigned u = __builtin_bit_cast(unsigned, f);
  unsigned r = (u + 0x7FFFu + ((u >> 16) & 1u)) >> 16;
  unsigned short s = (unsigned short)r;
  return __builtin_bit_cast(bf16_t, s);
}

// ---------------------------------------------------------------------------
// Gating: logits = x @ gate_w + gate_b; softmax; mask; normalize.
// ---------------------------------------------------------------------------
__global__ __launch_bounds__(256) void gate_kernel(
    const float* __restrict__ x, const float* __restrict__ gw,
    const float* __restrict__ gb, const float* __restrict__ mask,
    float* __restrict__ wout, float* __restrict__ out_mask,
    float* __restrict__ out_w) {
  const int lane = threadIdx.x & 31;
  const int wid  = threadIdx.x >> 5;
  const int row  = blockIdx.x * 8 + wid;
  float acc[NE_];
#pragma unroll
  for (int e = 0; e < NE_; ++e) acc[e] = 0.f;
  const float* xr = x + (size_t)row * D_;
  for (int d = lane; d < D_; d += 32) {
    float xv = xr[d];
#pragma unroll
    for (int e = 0; e < NE_; ++e) acc[e] += xv * gw[d * NE_ + e];
  }
#pragma unroll
  for (int off = 16; off > 0; off >>= 1)
#pragma unroll
    for (int e = 0; e < NE_; ++e) acc[e] += __shfl_xor(acc[e], off, 32);
  if (lane == 0) {
    float l[NE_], mk[NE_];
    float m = -3.4e38f;
#pragma unroll
    for (int e = 0; e < NE_; ++e) { l[e] = acc[e] + gb[e]; m = fmaxf(m, l[e]); }
    float s = 0.f;
#pragma unroll
    for (int e = 0; e < NE_; ++e) { l[e] = __expf(l[e] - m); s += l[e]; }
    float inv_s = 1.f / s, ms = 0.f;
#pragma unroll
    for (int e = 0; e < NE_; ++e) {
      mk[e] = mask[(size_t)row * NE_ + e];
      l[e] = l[e] * inv_s * mk[e];
      ms += l[e];
    }
    float inv = 1.f / (ms + 1e-9f);
#pragma unroll
    for (int e = 0; e < NE_; ++e) {
      float wv = l[e] * inv;
      wout[(size_t)row * NE_ + e]     = wv;
      out_w[(size_t)row * NE_ + e]    = wv;
      out_mask[(size_t)row * NE_ + e] = mk[e];
    }
  }
}

// ---------------------------------------------------------------------------
// x -> bf16 + cubic B-spline basis (uniform knots t(i)=0.4i-2.2), 8 per (b,d).
// ---------------------------------------------------------------------------
__global__ __launch_bounds__(256) void basis_kernel(
    const float* __restrict__ x, bf16_t* __restrict__ xb,
    bf16_t* __restrict__ basisb) {
  size_t tid = (size_t)blockIdx.x * 256 + threadIdx.x;  // over B_*D_
  float xv = x[tid];
  xb[tid] = f2bf(xv);
  float bb[11];
#pragma unroll
  for (int i = 0; i < 11; ++i) {
    float t0 = 0.4f * i - 2.2f;
    float t1 = t0 + 0.4f;
    bb[i] = (t0 <= xv && xv < t1) ? 1.f : 0.f;
  }
#pragma unroll
  for (int kk = 1; kk <= 3; ++kk) {
    float inv = 1.f / (0.4f * kk);
#pragma unroll
    for (int i = 0; i + kk < 11; ++i) {
      float ti   = 0.4f * i - 2.2f;
      float tik1 = 0.4f * (i + kk + 1) - 2.2f;
      bb[i] = (xv - ti) * inv * bb[i] + (tik1 - xv) * inv * bb[i + 1];
    }
  }
  bf16x8 v;
#pragma unroll
  for (int g = 0; g < 8; ++g) v[g] = f2bf(bb[g]);
  *(bf16x8*)&basisb[tid * 8] = v;
}

// ---------------------------------------------------------------------------
// Weight conversions: transposed bf16 copies (N-major, K contiguous).
// ---------------------------------------------------------------------------
__global__ __launch_bounds__(256) void conv_w1(const float* __restrict__ w,
                                               bf16_t* __restrict__ o) {
  size_t tid = (size_t)blockIdx.x * 256 + threadIdx.x;  // 4*H_*D_
  int d = (int)(tid & (D_ - 1));
  int h = (int)((tid >> 9) & (H_ - 1));
  int e = (int)(tid >> 20);
  o[tid] = f2bf(w[((size_t)e * D_ + d) * H_ + h]);
}

__global__ __launch_bounds__(256) void conv_w2(const float* __restrict__ w,
                                               bf16_t* __restrict__ o) {
  size_t tid = (size_t)blockIdx.x * 256 + threadIdx.x;  // 4*O_*H_
  int h  = (int)(tid & (H_ - 1));
  int oo = (int)((tid >> 11) & (O_ - 1));
  int e  = (int)(tid >> 20);
  o[tid] = f2bf(w[((size_t)e * H_ + h) * O_ + oo]);
}

__global__ __launch_bounds__(256) void conv_ceff(const float* __restrict__ co,
                                                 const float* __restrict__ sc,
                                                 bf16_t* __restrict__ o) {
  size_t tid = (size_t)blockIdx.x * 256 + threadIdx.x;  // 4*D_*O_
  int i  = (int)(tid & (D_ - 1));
  int oo = (int)((tid >> 9) & (O_ - 1));
  int e  = (int)(tid >> 18);
  size_t rb = (((size_t)e * D_ + i) * O_ + oo) * 8;
  float s = sc[((size_t)e * D_ + i) * O_ + oo];
  size_t wb = ((size_t)e * O_ + oo) * (size_t)(D_ * 8) + (size_t)i * 8;
  bf16x8 v;
#pragma unroll
  for (int g = 0; g < 8; ++g) v[g] = f2bf(co[rb + g] * s);
  *(bf16x8*)&o[wb] = v;
}

// ---------------------------------------------------------------------------
// TDM issue: one D# describing a [rows x 32] bf16 tile of a row-major matrix
// with leading dimension ldk (elements), DMA'd into LDS at ldsOff.
// 6-arg builtin form (amdgpu-toolchain / clang-23): groups 2/3 + trailing
// group are zero (2-D tensor, no iterate/gather).
// ---------------------------------------------------------------------------
#if defined(USE_TDM)
__device__ __forceinline__ void tdm_tile_load(const bf16_t* g, int rowOff,
                                              int ldk, int mTot, int rows,
                                              unsigned ldsOff, int kb) {
  unsigned long long ga =
      (unsigned long long)(const void*)g +
      ((unsigned long long)rowOff * (unsigned)ldk + (unsigned)kb) * 2ull;
  u32x4 g0;
  g0.x = 1u;                                   // count=1, user mode
  g0.y = ldsOff;                               // lds_addr
  g0.z = (unsigned)ga;                         // global_addr[31:0]
  g0.w = (unsigned)((ga >> 32) & 0x1FFFFFFu) | (2u << 30);  // addr hi | type=2
  i32x8 g1;
  g1[0] = 1 << 16;                             // data_size = 2 bytes
  g1[1] = (ldk & 0xFFFF) << 16;                // tensor_dim0[15:0]
  g1[2] = (int)(((unsigned)ldk >> 16) | (((unsigned)mTot & 0xFFFFu) << 16));
  g1[3] = (int)(((unsigned)mTot >> 16) | (32u << 16));  // tile_dim0 = 32
  g1[4] = rows;                                // tile_dim1 (tile_dim2=0)
  g1[5] = ldk;                                 // tensor_dim0_stride[31:0]
  g1[6] = 0;
  g1[7] = 0;
  i32x4 z4 = {0, 0, 0, 0};
  i32x8 z8 = {0, 0, 0, 0, 0, 0, 0, 0};
  __builtin_amdgcn_tensor_load_to_lds(g0, g1, z4, z4, z8, 0);
}
#endif

// ---------------------------------------------------------------------------
// Tiled bf16 WMMA GEMM:  C[M,N] = A[M,K] * Bt[N,K]^T   (both K-contiguous)
// Block tile (WR*MT*16) x (WC*NT*16), BK=32, 8 waves, wave tile MTx NT wmma.
// Double-buffered LDS; TDM fills tiles when available.
// MODE 0: +bias, ReLU, store bf16.  MODE 1: out += gatew[:,eidx] * (val+bias)
// ---------------------------------------------------------------------------
template <int MODE, int MT, int NT, int WR, int WC>
__global__ __launch_bounds__(256) void gemm_bf16(
    const bf16_t* __restrict__ A, const bf16_t* __restrict__ Bt,
    const float* __restrict__ bias, int M, int N, int Kd,
    bf16_t* __restrict__ Cb, float* __restrict__ out,
    const float* __restrict__ gatew, int eidx) {
  static_assert(WR * WC == 8, "8 waves per block");
  constexpr int BM = WR * MT * 16;
  constexpr int BN = WC * NT * 16;
  __shared__ __align__(16) bf16_t lA[2][BM * 32];
  __shared__ __align__(16) bf16_t lB[2][BN * 32];

  const int tid  = threadIdx.x;
  const int wid  = tid >> 5;
  const int lane = tid & 31;
  const int half = lane >> 4;
  const int l16  = lane & 15;
  const int wr   = wid % WR;
  const int wc   = wid / WR;
  const int rowBase = blockIdx.y * BM;
  const int colBase = blockIdx.x * BN;

  v8f acc[MT][NT];
#pragma unroll
  for (int mi = 0; mi < MT; ++mi)
#pragma unroll
    for (int ni = 0; ni < NT; ++ni) acc[mi][ni] = (v8f){0, 0, 0, 0, 0, 0, 0, 0};

  const int kTiles = Kd >> 5;

#if defined(USE_TDM)
  const unsigned ldsA0 = (unsigned)(unsigned long long)(void*)&lA[0][0];
  const unsigned ldsA1 = (unsigned)(unsigned long long)(void*)&lA[1][0];
  const unsigned ldsB0 = (unsigned)(unsigned long long)(void*)&lB[0][0];
  const unsigned ldsB1 = (unsigned)(unsigned long long)(void*)&lB[1][0];
  if (wid == 0) {
    tdm_tile_load(A, rowBase, Kd, M, BM, ldsA0, 0);
    tdm_tile_load(Bt, colBase, Kd, N, BN, ldsB0, 0);
  }
#else
  constexpr int ACH = BM / 64;  // 8-elt chunks per thread for A tile
  constexpr int BCH = BN / 64;
  bf16x8 ra[ACH], rb[BCH];
  {
#pragma unroll
    for (int c = 0; c < ACH; ++c) {
      int ch = c * 256 + tid, r = ch >> 2, kk = (ch & 3) << 3;
      ra[c] = *(const bf16x8*)&A[(size_t)(rowBase + r) * Kd + kk];
      *(bf16x8*)&lA[0][r * 32 + kk] = ra[c];
    }
#pragma unroll
    for (int c = 0; c < BCH; ++c) {
      int ch = c * 256 + tid, n = ch >> 2, kk = (ch & 3) << 3;
      rb[c] = *(const bf16x8*)&Bt[(size_t)(colBase + n) * Kd + kk];
      *(bf16x8*)&lB[0][n * 32 + kk] = rb[c];
    }
  }
#endif

  int cur = 0;
  for (int kt = 0; kt < kTiles; ++kt) {
#if defined(USE_TDM)
    if (wid == 0) __builtin_amdgcn_s_wait_tensorcnt(0);
    __syncthreads();  // buffer `cur` complete & previous reads drained
    if (wid == 0 && kt + 1 < kTiles) {
      int kb2 = (kt + 1) << 5;
      tdm_tile_load(A, rowBase, Kd, M, BM, cur ? ldsA0 : ldsA1, kb2);
      tdm_tile_load(Bt, colBase, Kd, N, BN, cur ? ldsB0 : ldsB1, kb2);
    }
#else
    const bool more = (kt + 1 < kTiles);
    if (more) {
      int kb2 = (kt + 1) << 5;
#pragma unroll
      for (int c = 0; c < ACH; ++c) {
        int ch = c * 256 + tid, r = ch >> 2, kk = (ch & 3) << 3;
        ra[c] = *(const bf16x8*)&A[(size_t)(rowBase + r) * Kd + kb2 + kk];
      }
#pragma unroll
      for (int c = 0; c < BCH; ++c) {
        int ch = c * 256 + tid, n = ch >> 2, kk = (ch & 3) << 3;
        rb[c] = *(const bf16x8*)&Bt[(size_t)(colBase + n) * Kd + kb2 + kk];
      }
      if (kt + 2 < kTiles) {
        __builtin_prefetch(&A[(size_t)(rowBase + (tid >> 1)) * Kd + ((kt + 2) << 5)], 0, 1);
        __builtin_prefetch(&Bt[(size_t)(colBase + (tid >> 2)) * Kd + ((kt + 2) << 5)], 0, 1);
      }
    }
    __syncthreads();  // buffer `cur` ready
#endif

    // ---- fragments (documented 16-bit layout): two ds_load_b128 each ----
    const int kf = half * 8;
    v16bf af[MT], bfr[NT];
#pragma unroll
    for (int mi = 0; mi < MT; ++mi) {
      int r = wr * (MT * 16) + mi * 16 + l16;
      bf16x8 lo = *(const bf16x8*)&lA[cur][r * 32 + kf];
      bf16x8 hi = *(const bf16x8*)&lA[cur][r * 32 + kf + 16];
      af[mi] = __builtin_shufflevector(lo, hi, 0, 1, 2, 3, 4, 5, 6, 7, 8, 9, 10,
                                       11, 12, 13, 14, 15);
    }
#pragma unroll
    for (int ni = 0; ni < NT; ++ni) {
      int n = wc * (NT * 16) + ni * 16 + l16;
      bf16x8 lo = *(const bf16x8*)&lB[cur][n * 32 + kf];
      bf16x8 hi = *(const bf16x8*)&lB[cur][n * 32 + kf + 16];
      bfr[ni] = __builtin_shufflevector(lo, hi, 0, 1, 2, 3, 4, 5, 6, 7, 8, 9, 10,
                                        11, 12, 13, 14, 15);
    }
#pragma unroll
    for (int mi = 0; mi < MT; ++mi)
#pragma unroll
      for (int ni = 0; ni < NT; ++ni)
        acc[mi][ni] = __builtin_amdgcn_wmma_f32_16x16x32_bf16(
            false, af[mi], false, bfr[ni], (short)0, acc[mi][ni], false, false);

#if !defined(USE_TDM)
    if (more) {
#pragma unroll
      for (int c = 0; c < ACH; ++c) {
        int ch = c * 256 + tid, r = ch >> 2, kk = (ch & 3) << 3;
        *(bf16x8*)&lA[cur ^ 1][r * 32 + kk] = ra[c];
      }
#pragma unroll
      for (int c = 0; c < BCH; ++c) {
        int ch = c * 256 + tid, n = ch >> 2, kk = (ch & 3) << 3;
        *(bf16x8*)&lB[cur ^ 1][n * 32 + kk] = rb[c];
      }
    }
#endif
    cur ^= 1;
  }

  // ---- epilogue: C lane map  Mrow = half*8 + r, Ncol = l16 ----
#pragma unroll
  for (int mi = 0; mi < MT; ++mi) {
#pragma unroll
    for (int ni = 0; ni < NT; ++ni) {
      int gcol = colBase + wc * (NT * 16) + ni * 16 + l16;
      float bv = bias ? bias[gcol] : 0.f;
#pragma unroll
      for (int r = 0; r < 8; ++r) {
        int grow = rowBase + wr * (MT * 16) + mi * 16 + half * 8 + r;
        float v = acc[mi][ni][r] + bv;
        if (MODE == 0) {
          v = v > 0.f ? v : 0.f;
          Cb[(size_t)grow * N + gcol] = f2bf(v);
        } else {
          float wv = gatew[(size_t)grow * NE_ + eidx];
          float* p = &out[(size_t)grow * N + gcol];
          *p += wv * v;  // unique per launch; launches serialize on stream
        }
      }
    }
  }
}

// ---------------------------------------------------------------------------
extern "C" void kernel_launch(void* const* d_in, const int* in_sizes, int n_in,
                              void* d_out, int out_size, void* d_ws,
                              size_t ws_size, hipStream_t stream) {
  const float* x      = (const float*)d_in[0];
  const float* emask  = (const float*)d_in[1];
  const float* gate_w = (const float*)d_in[2];
  const float* gate_b = (const float*)d_in[3];
  const float* mlp_w1 = (const float*)d_in[4];
  const float* mlp_b1 = (const float*)d_in[5];
  const float* mlp_w2 = (const float*)d_in[6];
  const float* mlp_b2 = (const float*)d_in[7];
  const float* kan_sc = (const float*)d_in[8];
  const float* kan_co = (const float*)d_in[9];

  float* out      = (float*)d_out;                  // [4096,512]
  float* out_mask = out + (size_t)B_ * O_;          // [4096,8]
  float* out_w    = out_mask + (size_t)B_ * NE_;    // [4096,8]

  char* ws = (char*)d_ws;
  auto alloc = [&](size_t bytes) {
    char* p = ws;
    ws += (bytes + 255) & ~(size_t)255;
    return p;
  };
  float*  ws_w   = (float*)alloc((size_t)B_ * NE_ * 4);
  bf16_t* xb     = (bf16_t*)alloc((size_t)B_ * D_ * 2);
  bf16_t* basisb = (bf16_t*)alloc((size_t)B_ * D_ * 8 * 2);      // [B, 4096]
  bf16_t* w1bT   = (bf16_t*)alloc((size_t)4 * H_ * D_ * 2);      // [e][h][d]
  bf16_t* w2bT   = (bf16_t*)alloc((size_t)4 * O_ * H_ * 2);      // [e][o][h]
  bf16_t* ceffT  = (bf16_t*)alloc((size_t)4 * O_ * D_ * 8 * 2);  // [e][o][i*8+g]
  bf16_t* hb     = (bf16_t*)alloc((size_t)B_ * H_ * 2);          // reused

  (void)hipMemsetAsync(d_out, 0, (size_t)B_ * O_ * sizeof(float), stream);

  gate_kernel<<<B_ / 8, 256, 0, stream>>>(x, gate_w, gate_b, emask, ws_w,
                                          out_mask, out_w);
  basis_kernel<<<(B_ * D_) / 256, 256, 0, stream>>>(x, xb, basisb);
  conv_w1<<<(4 * H_ * D_) / 256, 256, 0, stream>>>(mlp_w1, w1bT);
  conv_w2<<<(4 * O_ * H_) / 256, 256, 0, stream>>>(mlp_w2, w2bT);
  conv_ceff<<<(4 * D_ * O_) / 256, 256, 0, stream>>>(kan_co, kan_sc, ceffT);

  for (int e = 0; e < 4; ++e) {
    // 128x128 block, wave tile 64x32 (4x2 wmma)
    dim3 g1(H_ / 128, B_ / 128);
    gemm_bf16<0, 4, 2, 2, 4><<<g1, 256, 0, stream>>>(
        xb, w1bT + (size_t)e * H_ * D_, mlp_b1 + e * H_, B_, H_, D_, hb,
        nullptr, nullptr, 0);
    // 128x64 block, wave tile 32x32 (2x2 wmma)
    dim3 g2(O_ / 64, B_ / 128);
    gemm_bf16<1, 2, 2, 4, 2><<<g2, 256, 0, stream>>>(
        hb, w2bT + (size_t)e * O_ * H_, mlp_b2 + e * O_, B_, O_, H_, nullptr,
        out, ws_w, e);
  }
  for (int e = 0; e < 4; ++e) {
    dim3 g3(O_ / 64, B_ / 128);
    gemm_bf16<1, 2, 2, 4, 2><<<g3, 256, 0, stream>>>(
        basisb, ceffT + (size_t)e * O_ * D_ * 8, nullptr, B_, O_, D_ * 8,
        nullptr, out, ws_w, 4 + e);
  }
}
